// GraphHeterogenousCrossAttention_65438121722044
// MI455X (gfx1250) — matrix operand order repaired and checked by reading, not compiled, not run on hardware
//
#include <hip/hip_runtime.h>
#include <hip/hip_bf16.h>

typedef __attribute__((ext_vector_type(16))) __bf16 bf16x16;
typedef __attribute__((ext_vector_type(8)))  __bf16 bf16x8;
typedef __attribute__((ext_vector_type(8)))  float  f32x8;
typedef __attribute__((ext_vector_type(4)))  float  f32x4;

#define MROWS 65536   // B*N = 4*16384
#define DN    128
#define LDIM  256
#define HEADS 8
#define DH    32

// workspace layout (bytes)
#define WS_WFUSEDT 0        // bf16 [256][128]  (Wfused = W_node@Wq, transposed)
#define WS_WOUTT   65536    // bf16 [256][256]  (W_out transposed)
#define WS_K       196608   // f32  [256]
#define WS_VW      197632   // f32  [8][256]
#define WS_BFUSED  205824   // f32  [256]

// LDS layout (bf16 elements), padded pitches to kill bank conflicts:
// row stride in dwords % 64 == 4 -> 16 lanes hit 16 distinct banks.
#define WF_PITCH 136                       // 128 + 8
#define WO_PITCH 264                       // 256 + 8
#define Q_PITCH  264
#define SMEM_WF_OFF 0                      // 256*136 = 34816 elems (q tile reuses: 128*264=33792)
#define SMEM_WO_OFF (256 * WF_PITCH)
#define SMEM_ELEMS  (SMEM_WO_OFF + 256 * WO_PITCH)   // 102400 elems
#define SMEM_BYTES  (SMEM_ELEMS * 2)                 // 204800 B  (<= 320KB/WGP)

#if __has_builtin(__builtin_amdgcn_global_load_async_to_lds_b128) && \
    __has_builtin(__builtin_amdgcn_s_wait_asynccnt)
#define USE_ASYNC_LDS 1
#else
#define USE_ASYNC_LDS 0
#endif

// Builtin parameter types: pointer-to-int4 (gcc vector_size) in AS1 / AS3.
typedef int v4i_vs __attribute__((vector_size(16)));
typedef __attribute__((address_space(1))) v4i_vs as1_v4i;
typedef __attribute__((address_space(3))) v4i_vs as3_v4i;

// 16B global -> LDS copy (async DMA path on CDNA5, tracked by ASYNCcnt)
__device__ __forceinline__ void cp16_g2l(const __bf16* g, __bf16* l) {
#if USE_ASYNC_LDS
  __builtin_amdgcn_global_load_async_to_lds_b128(
      (as1_v4i*)(void*)g, (as3_v4i*)(void*)l, 0, 0);
#else
  *(bf16x8*)l = *(const bf16x8*)g;
#endif
}

__device__ __forceinline__ void cp_wait_async() {
#if USE_ASYNC_LDS
  __builtin_amdgcn_s_wait_asynccnt(0);
#endif
}

__device__ __forceinline__ bf16x16 pack2(bf16x8 a, bf16x8 b) {
  bf16x16 r;
  #pragma unroll
  for (int i = 0; i < 8; ++i) { r[i] = a[i]; r[8 + i] = b[i]; }
  return r;
}

// B fragment: lane holds column nt*16+lanelo, K at half*8 + {0..7} and 16 + half*8 + {0..7}
__device__ __forceinline__ bf16x16 ldbfrag(const __bf16* base, int pitch, int nt,
                                           int lanelo, int half, int kb) {
  const __bf16* p = base + (nt * 16 + lanelo) * pitch + kb + half * 8;
  return pack2(*(const bf16x8*)p, *(const bf16x8*)(p + 16));
}

__device__ __forceinline__ f32x8 wmma_bf16(bf16x16 a, bf16x16 b, f32x8 c) {
  return __builtin_amdgcn_wmma_f32_16x16x32_bf16(false, a, false, b, (short)0, c,
                                                 false, false);
}

// ---------------------------------------------------------------------------
// Kernel A0: WfusedT[n][i] = sum_j W_node[i][j] * Wq[j][n]   (bf16 out)
// ---------------------------------------------------------------------------
__global__ void prep_wfused(const float* __restrict__ W_node,
                            const float* __restrict__ Wq,
                            __bf16* __restrict__ WfT) {
  __shared__ float wqcol[LDIM];
  const int n = blockIdx.x;
  const int i = threadIdx.x;
  for (int j = threadIdx.x; j < LDIM; j += blockDim.x) wqcol[j] = Wq[j * LDIM + n];
  __syncthreads();
  const float* wrow = W_node + i * LDIM;
  float s = 0.f;
  #pragma unroll 4
  for (int j = 0; j < LDIM; ++j) s += wrow[j] * wqcol[j];
  WfT[n * DN + i] = (__bf16)s;
}

// ---------------------------------------------------------------------------
// Kernel A1: graph-path constants (k, vW, bfused, W_outT)
// ---------------------------------------------------------------------------
__global__ void prep_small(const float* __restrict__ energy,
                           const float* __restrict__ W_graph, const float* __restrict__ b_graph,
                           const float* __restrict__ Wq, const float* __restrict__ bq,
                           const float* __restrict__ b_node,
                           const float* __restrict__ Wk, const float* __restrict__ bk,
                           const float* __restrict__ Wv, const float* __restrict__ bv,
                           const float* __restrict__ W_out,
                           float* __restrict__ k_out, float* __restrict__ vW,
                           float* __restrict__ bfused, __bf16* __restrict__ WoT) {
  __shared__ float lg[LDIM], kl[LDIM], vv[LDIM], red[2 * HEADS];
  const int t = threadIdx.x;  // 0..255
  float a = b_graph[t];
  #pragma unroll
  for (int d = 0; d < 32; ++d) a += energy[d] * W_graph[d * LDIM + t];
  lg[t] = a;
  __syncthreads();
  const float* Wk2 = Wk + 2 * LDIM * LDIM;
  const float* Wv2 = Wv + 2 * LDIM * LDIM;
  float ka = bk[2 * LDIM + t], va = bv[2 * LDIM + t];
  for (int i = 0; i < LDIM; ++i) {
    const float l = lg[i];
    ka += l * Wk2[i * LDIM + t];
    va += l * Wv2[i * LDIM + t];
  }
  kl[t] = ka; vv[t] = va;
  __syncthreads();
  if (t < HEADS) {
    float m = -1e30f;
    for (int d = 0; d < DH; ++d) m = fmaxf(m, kl[t * DH + d]);
    red[t] = m;
  }
  __syncthreads();
  const float e = __expf(ka - red[t >> 5]);
  kl[t] = e;
  __syncthreads();
  if (t < HEADS) {
    float s = 0.f;
    for (int d = 0; d < DH; ++d) s += kl[t * DH + d];
    red[HEADS + t] = s;
  }
  __syncthreads();
  k_out[t] = e / red[HEADS + (t >> 5)];
  #pragma unroll
  for (int h = 0; h < HEADS; ++h) {
    float s = 0.f;
    for (int d = 0; d < DH; ++d) s += vv[h * DH + d] * W_out[(h * DH + d) * LDIM + t];
    vW[h * LDIM + t] = s;
  }
  float bfv = bq[t];
  for (int i = 0; i < LDIM; ++i) bfv += b_node[i] * Wq[i * LDIM + t];
  bfused[t] = bfv;
  for (int kk = 0; kk < LDIM; ++kk) WoT[t * LDIM + kk] = (__bf16)W_out[kk * LDIM + t];
}

// ---------------------------------------------------------------------------
// Main fused kernel: 512 blocks x 256 threads (8 wave32), 128 rows/block.
//   P1: Wfused^T (64KB) + W_out^T (128KB) -> LDS via async global->LDS DMA
//   P2: GEMM1 logits = node @ Wfused      (WMMA bf16, K=128, B from LDS)
//   P3: per-head softmax (shuffles) + factor; q(bf16) -> LDS (reuses WF region)
//   P4: GEMM2 acc2 = q @ W_out            (WMMA bf16, K=256, A+B from LDS)
//   P5: out = acc2 + b_out + sum_h factor_h * vW[h]
// ---------------------------------------------------------------------------
__global__ void __launch_bounds__(256)
main_fused(const float* __restrict__ node,
           const __bf16* __restrict__ WfT,    // [256][128]
           const __bf16* __restrict__ WoT,    // [256][256]
           const float* __restrict__ kvec,
           const float* __restrict__ vW,
           const float* __restrict__ bfused,
           const float* __restrict__ b_out,
           float* __restrict__ out) {
  extern __shared__ __bf16 smem[];  // SMEM_BYTES dynamic

  const int tid    = threadIdx.x;
  const int wave   = tid >> 5;
  const int lane   = tid & 31;
  const int lanelo = lane & 15;
  const int half   = lane >> 4;
  const int mBase  = blockIdx.x * 128;

  // ---- Phase 1: stage both weight tiles into LDS (pitched) ----
  {
    // WfusedT: 256 rows x 128 bf16 = 4096 x 16B chunks
    for (int c = tid; c < 4096; c += 256) {
      const int row = c >> 4, cc = c & 15;
      cp16_g2l(WfT + row * DN + cc * 8, smem + SMEM_WF_OFF + row * WF_PITCH + cc * 8);
    }
    // W_outT: 256 rows x 256 bf16 = 8192 x 16B chunks
    for (int c = tid; c < 8192; c += 256) {
      const int row = c >> 5, cc = c & 31;
      cp16_g2l(WoT + row * LDIM + cc * 8, smem + SMEM_WO_OFF + row * WO_PITCH + cc * 8);
    }
    cp_wait_async();
  }
  __syncthreads();

  // ---- Phase 2: GEMM1 (A resident in regs, B pipelined from LDS) ----
  const int row = mBase + wave * 16 + lanelo;   // A layout: M = lanelo for both halves
  bf16x16 afr[4];
  {
    const float* arow = node + (size_t)row * DN;
    #pragma unroll
    for (int kt = 0; kt < 4; ++kt) {
      const float* ap = arow + kt * 32 + half * 8;
      const f32x4 a0 = *(const f32x4*)(ap);
      const f32x4 a1 = *(const f32x4*)(ap + 4);
      const f32x4 a2 = *(const f32x4*)(ap + 16);
      const f32x4 a3 = *(const f32x4*)(ap + 20);
      #pragma unroll
      for (int i = 0; i < 4; ++i) {
        afr[kt][i]      = (__bf16)a0[i];
        afr[kt][4 + i]  = (__bf16)a1[i];
        afr[kt][8 + i]  = (__bf16)a2[i];
        afr[kt][12 + i] = (__bf16)a3[i];
      }
    }
  }

  f32x8 acc[16];
  #pragma unroll
  for (int i = 0; i < 16; ++i) {
    #pragma unroll
    for (int j = 0; j < 8; ++j) acc[i][j] = 0.f;
  }

  const __bf16* wf = smem + SMEM_WF_OFF;
  #pragma unroll
  for (int kt = 0; kt < 4; ++kt) {
    const int kb = kt * 32;
    bf16x16 bcur = ldbfrag(wf, WF_PITCH, 0, lanelo, half, kb);
    #pragma unroll
    for (int nt = 0; nt < 16; ++nt) {
      bf16x16 bnxt = bcur;
      if (nt < 15) bnxt = ldbfrag(wf, WF_PITCH, nt + 1, lanelo, half, kb);
      acc[nt] = wmma_bf16(afr[kt], bcur, acc[nt]);
      bcur = bnxt;
    }
  }
  __syncthreads();  // everyone done reading Wfused region

  // ---- Phase 3: per-head softmax over DH=32 + factor; q -> LDS ----
  float fact[8][8];
  #pragma unroll
  for (int h = 0; h < 8; ++h) {
    const float kr0 = kvec[h * 32 + lanelo];
    const float kr1 = kvec[h * 32 + 16 + lanelo];
    const float bf0 = bfused[h * 32 + lanelo];
    const float bf1 = bfused[h * 32 + 16 + lanelo];
    #pragma unroll
    for (int r = 0; r < 8; ++r) {
      float x0 = acc[2 * h][r] + bf0;
      float x1 = acc[2 * h + 1][r] + bf1;
      float m = fmaxf(x0, x1);
      for (int off = 1; off < 16; off <<= 1) m = fmaxf(m, __shfl_xor(m, off, 32));
      const float e0 = __expf(x0 - m);
      const float e1 = __expf(x1 - m);
      float s = e0 + e1;
      for (int off = 1; off < 16; off <<= 1) s += __shfl_xor(s, off, 32);
      const float inv = 1.0f / s;
      const float q0 = e0 * inv, q1 = e1 * inv;
      float d = q0 * kr0 + q1 * kr1;
      for (int off = 1; off < 16; off <<= 1) d += __shfl_xor(d, off, 32);
      fact[h][r] = d / fmaxf(d, 1e-8f);          // (q.k) * 1/clip(q.k, 1e-8)
      const int lrow = wave * 16 + r + half * 8;
      smem[lrow * Q_PITCH + h * 32 + lanelo]      = (__bf16)q0;
      smem[lrow * Q_PITCH + h * 32 + 16 + lanelo] = (__bf16)q1;
    }
  }
  __syncthreads();  // q tile ready

  // ---- Phase 4: GEMM2 (A = q tile LDS, B = W_outT LDS, pipelined) ----
  f32x8 acc2[16];
  #pragma unroll
  for (int i = 0; i < 16; ++i) {
    #pragma unroll
    for (int j = 0; j < 8; ++j) acc2[i][j] = 0.f;
  }
  const __bf16* qt = smem;
  const __bf16* wo = smem + SMEM_WO_OFF;
  #pragma unroll
  for (int kt = 0; kt < 8; ++kt) {
    const int kb = kt * 32;
    const __bf16* ap = qt + (wave * 16 + lanelo) * Q_PITCH + kb + half * 8;
    const bf16x16 af = pack2(*(const bf16x8*)ap, *(const bf16x8*)(ap + 16));
    bf16x16 bcur = ldbfrag(wo, WO_PITCH, 0, lanelo, half, kb);
    #pragma unroll
    for (int nt = 0; nt < 16; ++nt) {
      bf16x16 bnxt = bcur;
      if (nt < 15) bnxt = ldbfrag(wo, WO_PITCH, nt + 1, lanelo, half, kb);
      acc2[nt] = wmma_bf16(af, bcur, acc2[nt]);
      bcur = bnxt;
    }
  }

  // ---- Phase 5: epilogue ----
  #pragma unroll
  for (int nt = 0; nt < 16; ++nt) {
    const int col = nt * 16 + lanelo;
    const float bo = b_out[col];
    float vwc[8];
    #pragma unroll
    for (int h = 0; h < 8; ++h) vwc[h] = vW[h * LDIM + col];
    #pragma unroll
    for (int r = 0; r < 8; ++r) {
      float o = acc2[nt][r] + bo;
      #pragma unroll
      for (int h = 0; h < 8; ++h) o += fact[h][r] * vwc[h];
      const int orow = mBase + wave * 16 + r + half * 8;
      out[(size_t)orow * LDIM + col] = o;   // two 64B segments per store
    }
  }
}

// ---------------------------------------------------------------------------
extern "C" void kernel_launch(void* const* d_in, const int* in_sizes, int n_in,
                              void* d_out, int out_size, void* d_ws, size_t ws_size,
                              hipStream_t stream) {
  const float* node    = (const float*)d_in[0];
  // d_in[1] edge_features: dead (reference overwrites `out`; only graph branch survives)
  const float* energy  = (const float*)d_in[2];
  const float* W_node  = (const float*)d_in[3];
  const float* b_node  = (const float*)d_in[4];
  const float* W_graph = (const float*)d_in[7];
  const float* b_graph = (const float*)d_in[8];
  const float* Wq      = (const float*)d_in[9];
  const float* bq      = (const float*)d_in[10];
  const float* Wk      = (const float*)d_in[11];
  const float* bk      = (const float*)d_in[12];
  const float* Wv      = (const float*)d_in[13];
  const float* bv      = (const float*)d_in[14];
  const float* W_out   = (const float*)d_in[15];
  const float* b_out   = (const float*)d_in[16];

  char* ws = (char*)d_ws;
  __bf16* WfT  = (__bf16*)(ws + WS_WFUSEDT);
  __bf16* WoT  = (__bf16*)(ws + WS_WOUTT);
  float* kbuf  = (float*)(ws + WS_K);
  float* vWbuf = (float*)(ws + WS_VW);
  float* bfbuf = (float*)(ws + WS_BFUSED);
  float* out   = (float*)d_out;

  prep_wfused<<<256, 128, 0, stream>>>(W_node, Wq, WfT);
  prep_small<<<1, 256, 0, stream>>>(energy, W_graph, b_graph, Wq, bq, b_node,
                                    Wk, bk, Wv, bv, W_out, kbuf, vWbuf, bfbuf, WoT);
  main_fused<<<MROWS / 128, 256, SMEM_BYTES, stream>>>(node, WfT, WoT, kbuf, vWbuf,
                                                       bfbuf, b_out, out);
}
